// NeuronMLPBlock_6691559047325
// MI455X (gfx1250) — matrix operand order repaired and checked by reading, not compiled, not run on hardware
//
#include <hip/hip_runtime.h>
#include <hip/hip_bf16.h>

// ---------------------------------------------------------------------------
// MoE SwiGLU block for MI455X (gfx1250): bf16 WMMA + double-buffered TDM
// gather + routed experts. M=32 token tiles, 8 waves x 16 N-cols per block.
// ---------------------------------------------------------------------------

typedef __attribute__((ext_vector_type(16))) __bf16 v16bf;
typedef __attribute__((ext_vector_type(8)))  __bf16 v8bf;
typedef __attribute__((ext_vector_type(8)))  float  v8f;
typedef __attribute__((ext_vector_type(4)))  unsigned int u32x4;
typedef __attribute__((ext_vector_type(8)))  int i32x8;
typedef __attribute__((ext_vector_type(4)))  int i32x4;

static constexpr int T_TOK = 4096;   // B*S tokens
static constexpr int HDIM  = 1024;
static constexpr int IDIM  = 2816;
static constexpr int EXP   = 8;
static constexpr int KC    = 256;    // K chunk staged in LDS per TDM op
static constexpr int KCP   = KC + 8; // padded LDS row stride (TDM pad: +4 dwords/row)
static constexpr int MT    = 32;     // token tile rows per block

// ---------------- WMMA helpers (CDNA5 16x16x32 bf16, f32 accum) -------------

__device__ __forceinline__ v8f wmma_bf16(v16bf a, v16bf b, v8f c) {
  return __builtin_amdgcn_wmma_f32_16x16x32_bf16(false, a, false, b,
                                                 (short)0, c, false, false);
}

// A fragment (16x32 M x K) from LDS tile, padded row stride KCP elements.
// Lane L: row = L%16, 16 contiguous K at base=(L/16)*16
__device__ __forceinline__ v16bf load_a_lds(const __bf16* lds, int lane, int kk) {
  int r = lane & 15, half = lane >> 4;
  const __bf16* p = lds + r * KCP + kk + half * 16;
  union { v16bf v; v8bf h[2]; } u;
  u.h[0] = *(const v8bf*)(p);
  u.h[1] = *(const v8bf*)(p + 8);
  return u.v;
}

// B fragment (32x16 K x N) from column-major (transposed) weights:
// lane L holds column c = L%16, 16 contiguous K at kbase=(L/16)*16.
__device__ __forceinline__ v16bf load_b(const __bf16* Wt, int lane,
                                        long col_stride, int k) {
  int c = lane & 15, half = lane >> 4;
  return *(const v16bf*)(Wt + (long)c * col_stride + k + half * 16);
}

// ---------------- TDM descriptor issue (tensor_load_to_lds) ----------------

__device__ __forceinline__ int rfl(int x) {
  return __builtin_amdgcn_readfirstlane(x);
}

// Issue one TDM load of an [nrows x KC] bf16 tile into LDS (no wait).
// LDS rows are padded by 4 dwords (pad_interval=6: every 128 dwords,
// pad_amount=3: +4 dwords) -> row stride KCP elements, bank-conflict free.
// gather: rows picked by 16-bit indices pk[0..7] (2/dword), else consecutive.
__device__ __forceinline__ void tdm_load_tile(unsigned ldsAddr,
                                              unsigned long long gaddr,
                                              int srcRowStride,
                                              int nrows, int tensorRows,
                                              bool gather, const int* pk) {
  u32x4 g0;
  g0[0] = (unsigned)rfl((int)((gather ? (1u << 31) : 0u) | 1u)); // gather_mode | count=1
  g0[1] = (unsigned)rfl((int)ldsAddr);
  g0[2] = (unsigned)rfl((int)(unsigned)(gaddr & 0xFFFFFFFFull));
  g0[3] = (unsigned)rfl((int)((unsigned)((gaddr >> 32) & 0x1FFFFFFull) | (2u << 30)));

  i32x8 g1;
  // data_size=2B | pad_enable | pad_interval=6 (128 dw) | pad_amount=3 (4 dw)
  g1[0] = rfl((1 << 16) | (1 << 20) | (6 << 22) | (3 << 25));
  g1[1] = rfl((KC & 0xFFFF) << 16);                              // tensor_dim0 lo
  g1[2] = rfl((tensorRows & 0xFFFF) << 16);                      // td0 hi=0 | td1 lo
  g1[3] = rfl(((tensorRows >> 16) & 0xFFFF) | (KC << 16));       // td1 hi | tile_dim0
  g1[4] = rfl(nrows & 0xFFFF);                                   // tile_dim1 | tile_dim2=0
  g1[5] = rfl(srcRowStride);                                     // stride0 lo32
  g1[6] = rfl(0);
  g1[7] = rfl(0);

  i32x4 g2, g3;
  if (gather) {
    g2[0] = rfl(pk[0]); g2[1] = rfl(pk[1]); g2[2] = rfl(pk[2]); g2[3] = rfl(pk[3]);
    g3[0] = rfl(pk[4]); g3[1] = rfl(pk[5]); g3[2] = rfl(pk[6]); g3[3] = rfl(pk[7]);
  } else {
    g2[0] = g2[1] = g2[2] = g2[3] = 0;
    g3[0] = g3[1] = g3[2] = g3[3] = 0;
  }

#if defined(__clang_major__) && (__clang_major__ >= 23)
  i32x8 g4; g4[0]=g4[1]=g4[2]=g4[3]=g4[4]=g4[5]=g4[6]=g4[7]=0;
  __builtin_amdgcn_tensor_load_to_lds(g0, g1, g2, g3, g4, 0);
#else
  __builtin_amdgcn_tensor_load_to_lds(g0, g1, g2, g3, 0);
#endif
}

// ---------------- Kernel 0: f32 -> bf16 copy (activations) -----------------

__global__ __launch_bounds__(256)
void convert_x_kernel(const float* __restrict__ in, __bf16* __restrict__ out, int n) {
  for (int i = blockIdx.x * blockDim.x + threadIdx.x; i < n; i += gridDim.x * blockDim.x)
    out[i] = (__bf16)in[i];
}

// ------- Kernel 0b: f32 [E][R][C] -> bf16 [E][C][R] LDS-tiled transpose ----

__global__ __launch_bounds__(256)
void transpose_conv_kernel(const float* __restrict__ in, __bf16* __restrict__ out,
                           int R, int C) {
  __shared__ float tile[32][33];
  size_t eoff = (size_t)blockIdx.z * (size_t)R * (size_t)C;
  int c0 = blockIdx.x * 32, r0 = blockIdx.y * 32;
  for (int j = threadIdx.y; j < 32; j += 8)
    tile[j][threadIdx.x] = in[eoff + (size_t)(r0 + j) * C + c0 + threadIdx.x];
  __syncthreads();
  for (int j = threadIdx.y; j < 32; j += 8)
    out[eoff + (size_t)(c0 + j) * R + r0 + threadIdx.x] = (__bf16)tile[threadIdx.x][j];
}

// ---------------- Kernel 1: router (one wave32 per token) ------------------

__global__ __launch_bounds__(128)
void router_kernel(const float* __restrict__ x, const float* __restrict__ wr,
                   int* __restrict__ counts, int* __restrict__ idxList,
                   float* __restrict__ wgtList) {
  int wave = (int)((blockIdx.x * blockDim.x + threadIdx.x) >> 5);
  int lane = threadIdx.x & 31;
  if (wave >= T_TOK) return;
  float acc[EXP] = {};
  const float* xr = x + (size_t)wave * HDIM;
  for (int h = lane; h < HDIM; h += 32) {
    float xv = xr[h];
    const float* w = wr + (size_t)h * EXP;
    #pragma unroll
    for (int e = 0; e < EXP; ++e) acc[e] += xv * w[e];
  }
  #pragma unroll
  for (int e = 0; e < EXP; ++e)
    #pragma unroll
    for (int off = 16; off > 0; off >>= 1)
      acc[e] += __shfl_down(acc[e], off, 32);
  if (lane == 0) {
    float mx = acc[0];
    #pragma unroll
    for (int e = 1; e < EXP; ++e) mx = fmaxf(mx, acc[e]);
    float p[EXP];
    #pragma unroll
    for (int e = 0; e < EXP; ++e) p[e] = __expf(acc[e] - mx);
    int i0 = 0;
    #pragma unroll
    for (int e = 1; e < EXP; ++e) if (p[e] > p[i0]) i0 = e;
    int i1 = (i0 == 0) ? 1 : 0;
    #pragma unroll
    for (int e = 0; e < EXP; ++e) if (e != i0 && p[e] > p[i1]) i1 = e;
    float inv = 1.0f / (p[i0] + p[i1]);   // softmax denom cancels in renorm
    int s0 = atomicAdd(&counts[i0], 1);
    idxList[i0 * T_TOK + s0] = wave; wgtList[i0 * T_TOK + s0] = p[i0] * inv;
    int s1 = atomicAdd(&counts[i1], 1);
    idxList[i1 * T_TOK + s1] = wave; wgtList[i1 * T_TOK + s1] = p[i1] * inv;
  }
}

// ---------------- Kernel 2: 32-padded prefix offsets -----------------------

__global__ void scan_kernel(const int* __restrict__ counts, int* __restrict__ offsets) {
  if (threadIdx.x == 0 && blockIdx.x == 0) {
    int o = 0;
    for (int e = 0; e < EXP; ++e) { offsets[e] = o; o += (counts[e] + MT - 1) & ~(MT - 1); }
    offsets[EXP] = o;
  }
}

// ------- Kernel 3: GEMM1  h = silu(Xe @ W1t) * (Xe @ W3t)  -> Hbuf ---------
// grid: (IDIM/128, T/32, EXP), block 256 (8 waves, 16 N-cols each, M=32)

__global__ __launch_bounds__(256)
void gemm1_kernel(const __bf16* __restrict__ Xb,
                  const __bf16* __restrict__ W1T, const __bf16* __restrict__ W3T,
                  __bf16* __restrict__ Hb,
                  const int* __restrict__ idxList,
                  const int* __restrict__ counts, const int* __restrict__ offsets) {
  int e    = blockIdx.z;
  int tile = blockIdx.y;                       // 32-token tile
  int count = counts[e];
  if (tile * MT >= count) return;
  int slotBase = offsets[e] + tile * MT;
  int waveId = threadIdx.x >> 5;
  int lane   = threadIdx.x & 31;
  int n0     = (blockIdx.x * 8 + waveId) * 16;

  __shared__ __bf16 lAs[2][MT * KCP];          // double-buffered A tiles

  // Pack 32 token indices (16-bit each) for the two TDM gathers; wave-uniform.
  int pk[16];
  #pragma unroll
  for (int j = 0; j < 16; ++j) {
    int s0 = tile * MT + 2 * j, s1 = s0 + 1;
    unsigned a = (s0 < count) ? (unsigned)idxList[e * T_TOK + s0] : 0u;
    unsigned b = (s1 < count) ? (unsigned)idxList[e * T_TOK + s1] : 0u;
    pk[j] = (int)(a | (b << 16));
  }

  const __bf16* B1 = W1T + (size_t)e * IDIM * HDIM + (size_t)n0 * HDIM;
  const __bf16* B3 = W3T + (size_t)e * IDIM * HDIM + (size_t)n0 * HDIM;

  v8f accG0 = {}, accG1 = {}, accU0 = {}, accU1 = {};

  const int nchunks = HDIM / KC;
  if (waveId == 0) {  // preload chunk 0 (two 16-row gathers)
    unsigned lb = (unsigned)(size_t)(&lAs[0][0]);
    unsigned long long ga = (unsigned long long)(size_t)Xb;
    tdm_load_tile(lb, ga, HDIM, 16, T_TOK, true, pk);
    tdm_load_tile(lb + 16 * KCP * 2, ga, HDIM, 16, T_TOK, true, pk + 8);
  }

  for (int c = 0; c < nchunks; ++c) {
    int cur = c & 1;
    if (waveId == 0) {
      if (c + 1 < nchunks) {   // stream next chunk while computing this one
        unsigned lb = (unsigned)(size_t)(&lAs[cur ^ 1][0]);
        unsigned long long ga = (unsigned long long)(size_t)Xb +
                                (unsigned long long)(c + 1) * KC * 2ull;
        tdm_load_tile(lb, ga, HDIM, 16, T_TOK, true, pk);
        tdm_load_tile(lb + 16 * KCP * 2, ga, HDIM, 16, T_TOK, true, pk + 8);
        __builtin_amdgcn_s_wait_tensorcnt(2);  // chunk c done (in-order per wave)
      } else {
        __builtin_amdgcn_s_wait_tensorcnt(0);
      }
    }
    __syncthreads();
    const __bf16* la = &lAs[cur][0];
    int k0 = c * KC;
    #pragma unroll
    for (int kk = 0; kk < KC; kk += 32) {
      v16bf a0 = load_a_lds(la, lane, kk);                 // rows 0..15
      v16bf a1 = load_a_lds(la + 16 * KCP, lane, kk);      // rows 16..31
      v16bf b1 = load_b(B1, lane, HDIM, k0 + kk);
      v16bf b3 = load_b(B3, lane, HDIM, k0 + kk);
      accG0 = wmma_bf16(a0, b1, accG0);
      accG1 = wmma_bf16(a1, b1, accG1);
      accU0 = wmma_bf16(a0, b3, accU0);
      accU1 = wmma_bf16(a1, b3, accU1);
    }
    __syncthreads();
  }

  // SwiGLU epilogue -> compact activation buffer (bf16)
  int col = lane & 15, halfl = lane >> 4;
  #pragma unroll
  for (int v = 0; v < 8; ++v) {
    int m = v + 8 * halfl;
    float g0 = accG0[v], u0 = accU0[v];
    float g1 = accG1[v], u1 = accU1[v];
    float h0 = (g0 / (1.0f + __expf(-g0))) * u0;
    float h1 = (g1 / (1.0f + __expf(-g1))) * u1;
    Hb[(size_t)(slotBase + m) * IDIM + n0 + col]      = (__bf16)h0;
    Hb[(size_t)(slotBase + m + 16) * IDIM + n0 + col] = (__bf16)h1;
  }
}

// ------- Kernel 4: GEMM2  y = Hbuf @ W2t, scatter out += w * y -------------
// grid: (HDIM/128, T/32, EXP), block 256

__global__ __launch_bounds__(256)
void gemm2_kernel(const __bf16* __restrict__ Hb, const __bf16* __restrict__ W2T,
                  float* __restrict__ out,
                  const int* __restrict__ idxList, const float* __restrict__ wgtList,
                  const int* __restrict__ counts, const int* __restrict__ offsets) {
  int e    = blockIdx.z;
  int tile = blockIdx.y;
  int count = counts[e];
  if (tile * MT >= count) return;
  int slotBase = offsets[e] + tile * MT;
  int waveId = threadIdx.x >> 5;
  int lane   = threadIdx.x & 31;
  int n0     = (blockIdx.x * 8 + waveId) * 16;

  __shared__ __bf16 lAs[2][MT * KCP];

  const __bf16* B2 = W2T + (size_t)e * HDIM * IDIM + (size_t)n0 * IDIM;

  v8f acc0 = {}, acc1 = {};

  const int nchunks = IDIM / KC;
  unsigned long long gbase = (unsigned long long)(size_t)Hb +
                             (unsigned long long)slotBase * IDIM * 2ull;
  if (waveId == 0) {   // preload chunk 0: one 32-row 2D tile
    tdm_load_tile((unsigned)(size_t)(&lAs[0][0]), gbase, IDIM, MT, MT, false, nullptr);
  }

  for (int c = 0; c < nchunks; ++c) {
    int cur = c & 1;
    if (waveId == 0) {
      if (c + 1 < nchunks) {
        tdm_load_tile((unsigned)(size_t)(&lAs[cur ^ 1][0]),
                      gbase + (unsigned long long)(c + 1) * KC * 2ull,
                      IDIM, MT, MT, false, nullptr);
        __builtin_amdgcn_s_wait_tensorcnt(1);
      } else {
        __builtin_amdgcn_s_wait_tensorcnt(0);
      }
    }
    __syncthreads();
    const __bf16* la = &lAs[cur][0];
    int k0 = c * KC;
    #pragma unroll
    for (int kk = 0; kk < KC; kk += 32) {
      v16bf a0 = load_a_lds(la, lane, kk);
      v16bf a1 = load_a_lds(la + 16 * KCP, lane, kk);
      v16bf b  = load_b(B2, lane, IDIM, k0 + kk);
      acc0 = wmma_bf16(a0, b, acc0);
      acc1 = wmma_bf16(a1, b, acc1);
    }
    __syncthreads();
  }

  // Weighted scatter into output (each token lives in exactly 2 expert lists)
  int col = lane & 15, halfl = lane >> 4;
  #pragma unroll
  for (int v = 0; v < 8; ++v) {
    int m = v + 8 * halfl;
    #pragma unroll
    for (int h2 = 0; h2 < 2; ++h2) {
      int slot = tile * MT + m + 16 * h2;
      if (slot < count) {
        int tok = idxList[e * T_TOK + slot];
        float w = wgtList[e * T_TOK + slot];
        float val = (h2 == 0) ? acc0[v] : acc1[v];
        atomicAdd(out + (size_t)tok * HDIM + n0 + col, w * val);
      }
    }
  }
}

// ---------------------------------------------------------------------------

extern "C" void kernel_launch(void* const* d_in, const int* in_sizes, int n_in,
                              void* d_out, int out_size, void* d_ws, size_t ws_size,
                              hipStream_t stream) {
  const float* x  = (const float*)d_in[0];   // [T, H]
  const float* wr = (const float*)d_in[1];   // [H, E]
  const float* w1 = (const float*)d_in[2];   // [E, H, I]
  const float* w2 = (const float*)d_in[3];   // [E, I, H]
  const float* w3 = (const float*)d_in[4];   // [E, H, I]
  float* out = (float*)d_out;

  char* ws = (char*)d_ws;
  size_t off = 0;
  auto alloc = [&](size_t bytes) -> void* {
    void* p = ws + off;
    off = (off + bytes + 255) & ~(size_t)255;
    return p;
  };
  const size_t HROWS = (size_t)2 * T_TOK + (size_t)MT * EXP;   // padded slots
  __bf16* Xb   = (__bf16*)alloc((size_t)T_TOK * HDIM * 2);
  __bf16* W1T  = (__bf16*)alloc((size_t)EXP * IDIM * HDIM * 2);
  __bf16* W3T  = (__bf16*)alloc((size_t)EXP * IDIM * HDIM * 2);
  __bf16* W2T  = (__bf16*)alloc((size_t)EXP * HDIM * IDIM * 2);
  __bf16* Hb   = (__bf16*)alloc(HROWS * IDIM * 2);
  int*    idxL = (int*)  alloc((size_t)EXP * T_TOK * 4);
  float*  wgtL = (float*)alloc((size_t)EXP * T_TOK * 4);
  int*    cnts = (int*)  alloc(EXP * 4);
  int*    offs = (int*)  alloc((EXP + 1) * 4);

  hipMemsetAsync(cnts, 0, EXP * 4, stream);
  hipMemsetAsync(out, 0, (size_t)T_TOK * HDIM * 4, stream);

  // bf16 conversion + weight transposes (column-major for WMMA B fragments)
  convert_x_kernel<<<2048, 256, 0, stream>>>(x, Xb, T_TOK * HDIM);
  transpose_conv_kernel<<<dim3(IDIM / 32, HDIM / 32, EXP), dim3(32, 8), 0, stream>>>(
      w1, W1T, HDIM, IDIM);   // [E][H][I] -> [E][I][H]
  transpose_conv_kernel<<<dim3(IDIM / 32, HDIM / 32, EXP), dim3(32, 8), 0, stream>>>(
      w3, W3T, HDIM, IDIM);
  transpose_conv_kernel<<<dim3(HDIM / 32, IDIM / 32, EXP), dim3(32, 8), 0, stream>>>(
      w2, W2T, IDIM, HDIM);   // [E][I][H] -> [E][H][I]

  router_kernel<<<T_TOK / 4, 128, 0, stream>>>(x, wr, cnts, idxL, wgtL);
  scan_kernel<<<1, 32, 0, stream>>>(cnts, offs);

  gemm1_kernel<<<dim3(IDIM / 128, T_TOK / MT, EXP), 256, 0, stream>>>(
      Xb, W1T, W3T, Hb, idxL, cnts, offs);
  gemm2_kernel<<<dim3(HDIM / 128, T_TOK / MT, EXP), 256, 0, stream>>>(
      Hb, W2T, out, idxL, wgtL, cnts, offs);
}
